// Attention_Layer_49203145343093
// MI455X (gfx1250) — matrix operand order repaired and checked by reading, not compiled, not run on hardware
//
#include <hip/hip_runtime.h>
#include <stdint.h>

#define HIDDEN 1024
#define SEQ    2048
#define BATCH  4
#define MPROJ  (BATCH * SEQ)   // 8192

typedef __attribute__((ext_vector_type(16))) __bf16    v16bf;
typedef __attribute__((ext_vector_type(8)))  __bf16    v8bf;
typedef __attribute__((ext_vector_type(8)))  float     v8f;
typedef __attribute__((ext_vector_type(8)))  unsigned  v8u;

// ---------------- async global->LDS path (device/gfx1250 only) ----------------
#if defined(__gfx1250__)
#define USE_ASYNC 1
#else
#define USE_ASYNC 0
#endif

// Per-lane async DMA: 16B global -> LDS, tracked by ASYNCcnt.
// VDST = LDS byte address (low 32 bits of generic pointer), VADDR = 64-bit
// global address, GV mode (saddr = off).  CDNA5 ISA.
__device__ __forceinline__ void async_copy16(const __bf16* g, __bf16* l) {
#if USE_ASYNC
  unsigned lds_off = (unsigned)(size_t)l;
  unsigned long long gaddr = (unsigned long long)(size_t)g;
  asm volatile("global_load_async_to_lds_b128 %0, %1, off"
               :
               : "v"(lds_off), "v"(gaddr)
               : "memory");
#else
  *(uint4*)l = *(const uint4*)g;
#endif
}

// newest batch of 6 async copies may stay in flight; older batch drained
__device__ __forceinline__ void wait_async_keep6() {
#if USE_ASYNC
  asm volatile("s_wait_asynccnt 0x6" ::: "memory");
#endif
}
__device__ __forceinline__ void wait_async_all() {
#if USE_ASYNC
  asm volatile("s_wait_asynccnt 0x0" ::: "memory");
#endif
}

__device__ __forceinline__ v8f wmma_bf16(v16bf a, v16bf b, v8f c) {
  return __builtin_amdgcn_wmma_f32_16x16x32_bf16(false, a, false, b,
                                                 (short)0, c, false, false);
}

// ---------------- kernel: elementwise f32 -> bf16 ----------------
__global__ __launch_bounds__(256) void convert_kernel(
    const float* __restrict__ src, __bf16* __restrict__ dst, int n8) {
  const int idx = blockIdx.x * 256 + threadIdx.x;   // 8 elements per thread
  if (idx < n8) {
    const float4* p = (const float4*)(src + (size_t)idx * 8);
    float4 a = p[0], b = p[1];
    v8f f = {a.x, a.y, a.z, a.w, b.x, b.y, b.z, b.w};
    *(v8bf*)(dst + (size_t)idx * 8) = __builtin_convertvector(f, v8bf);
  }
}

// ---------------- unified NT GEMM: C[M,N] = A[M,K] * B[N,K]^T ----------------
// Block: 256 threads = 8 waves (2 m-waves x 4 n-waves). Block tile 128x256.
// A,B panels (128x32 / 256x32 bf16) double-buffered in LDS via async copies.
// OUT_MODE: 0 = bf16 row-major (ldo)     1 = bf16, V-transposed [B,HIDDEN,SEQ]
//           2 = f32 row-major (ldo)
template <int OUT_MODE>
__global__ __launch_bounds__(256) void gemm_nt_kernel(
    const __bf16* __restrict__ A, const __bf16* __restrict__ Bm,
    void* __restrict__ Out, int K, int tiles_n, int ldo) {
  __shared__ __bf16 smA[2][128 * 32];   // 8 KB per buffer
  __shared__ __bf16 smB[2][256 * 32];   // 16 KB per buffer

  const int t    = threadIdx.x;
  const int lane = t & 31;
  const int w    = t >> 5;
  const int wm   = w >> 2;   // 0..1
  const int wn   = w & 3;    // 0..3
  const int mtile = (blockIdx.x / tiles_n) * 128;
  const int ntile = (blockIdx.x % tiles_n) * 256;

  // cooperative copy assignments
  const int arow = t >> 1;           // 0..127
  const int aoff = (t & 1) * 16;     // element offset in 32-wide K slab
  const __bf16* gA = A + (size_t)(mtile + arow) * K + aoff;
  const __bf16* gB = Bm + (size_t)(ntile + t) * K;

  auto load_panels = [&](int buf, int k0) {
    const __bf16* ga = gA + k0;
    __bf16* la = &smA[buf][arow * 32 + aoff];
    async_copy16(ga, la);
    async_copy16(ga + 8, la + 8);
    const __bf16* gb = gB + k0;
    __bf16* lb = &smB[buf][t * 32];
    async_copy16(gb, lb);
    async_copy16(gb + 8, lb + 8);
    async_copy16(gb + 16, lb + 16);
    async_copy16(gb + 24, lb + 24);
  };

  // fragment loads from LDS (contiguous k order k = 16*(lane>>4) + s for A and B, 16-bit A 16x32 / B 32x16)
  auto frag_a = [&](int buf, int i) -> v16bf {
    const __bf16* p =
        &smA[buf][(wm * 64 + 16 * i + (lane & 15)) * 32 + ((lane >> 4) << 4)];
    uint4 x = *(const uint4*)p;          // K .. K+7
    uint4 y = *(const uint4*)(p + 8);    // K+8 .. K+15
    v8u u = {x.x, x.y, x.z, x.w, y.x, y.y, y.z, y.w};
    return __builtin_bit_cast(v16bf, u);
  };
  auto frag_b = [&](int buf, int j) -> v16bf {
    const __bf16* p =
        &smB[buf][(wn * 64 + 16 * j + (lane & 15)) * 32 + ((lane >> 4) << 4)];
    v8u u = *(const v8u*)p;              // 16 contiguous K values (32B)
    return __builtin_bit_cast(v16bf, u);
  };

  v8f acc[4][4];
#pragma unroll
  for (int i = 0; i < 4; ++i)
#pragma unroll
    for (int j = 0; j < 4; ++j) acc[i][j] = (v8f)0.0f;

  const int nk = K >> 5;
  load_panels(0, 0);
  for (int kk = 0; kk < nk; ++kk) {
    if (kk + 1 < nk) {
      load_panels((kk + 1) & 1, (kk + 1) * 32);
      wait_async_keep6();
    } else {
      wait_async_all();
    }
    __syncthreads();

    const int buf = kk & 1;
    v16bf a[4], b[4];
#pragma unroll
    for (int i = 0; i < 4; ++i) a[i] = frag_a(buf, i);
#pragma unroll
    for (int j = 0; j < 4; ++j) b[j] = frag_b(buf, j);
#pragma unroll
    for (int i = 0; i < 4; ++i)
#pragma unroll
      for (int j = 0; j < 4; ++j) acc[i][j] = wmma_bf16(a[i], b[j], acc[i][j]);

    __syncthreads();
  }

  // epilogue: C layout row = vgpr + 8*(lane>=16), col = lane&15
#pragma unroll
  for (int i = 0; i < 4; ++i)
#pragma unroll
    for (int j = 0; j < 4; ++j) {
      const int col  = ntile + wn * 64 + 16 * j + (lane & 15);
      const int row0 = mtile + wm * 64 + 16 * i + ((lane >> 4) << 3);
      if (OUT_MODE == 1) {
        // V transposed: Vt[b][col][s], rows row0..row0+7 contiguous in s
        const int b = row0 >> 11;           // / SEQ
        const int s = row0 & (SEQ - 1);
        v8bf v = __builtin_convertvector(acc[i][j], v8bf);
        *(v8bf*)((__bf16*)Out + ((size_t)b * HIDDEN + col) * SEQ + s) = v;
      } else if (OUT_MODE == 0) {
#pragma unroll
        for (int r = 0; r < 8; ++r)
          ((__bf16*)Out)[(size_t)(row0 + r) * ldo + col] =
              (__bf16)acc[i][j][r];
      } else {
#pragma unroll
        for (int r = 0; r < 8; ++r)
          ((float*)Out)[(size_t)(row0 + r) * ldo + col] = acc[i][j][r];
      }
    }
}

// ---------------- kernel: row softmax, f32 scores -> bf16 probs ----------------
__global__ __launch_bounds__(256) void softmax_kernel(
    const float* __restrict__ Sc, __bf16* __restrict__ Pb) {
  __shared__ float red[256];
  const int row = blockIdx.x;
  const int tid = threadIdx.x;
  const float* p = Sc + (size_t)row * SEQ + tid * 8;
  float4 v0 = *(const float4*)(p);
  float4 v1 = *(const float4*)(p + 4);
  float f[8] = {v0.x, v0.y, v0.z, v0.w, v1.x, v1.y, v1.z, v1.w};

  float m = f[0];
#pragma unroll
  for (int e = 1; e < 8; ++e) m = fmaxf(m, f[e]);
  red[tid] = m;
  __syncthreads();
  for (int s = 128; s > 0; s >>= 1) {
    if (tid < s) red[tid] = fmaxf(red[tid], red[tid + s]);
    __syncthreads();
  }
  m = red[0];
  __syncthreads();

  float sum = 0.0f;
#pragma unroll
  for (int e = 0; e < 8; ++e) {
    f[e] = __expf(f[e] - m);
    sum += f[e];
  }
  red[tid] = sum;
  __syncthreads();
  for (int s = 128; s > 0; s >>= 1) {
    if (tid < s) red[tid] += red[tid + s];
    __syncthreads();
  }
  const float inv = 1.0f / red[0];

  v8f o = {f[0] * inv, f[1] * inv, f[2] * inv, f[3] * inv,
           f[4] * inv, f[5] * inv, f[6] * inv, f[7] * inv};
  *(v8bf*)(Pb + (size_t)row * SEQ + tid * 8) = __builtin_convertvector(o, v8bf);
}

// ---------------- launch ----------------
extern "C" void kernel_launch(void* const* d_in, const int* in_sizes, int n_in,
                              void* d_out, int out_size, void* d_ws,
                              size_t ws_size, hipStream_t stream) {
  const float* query = (const float*)d_in[0];
  const float* key   = (const float*)d_in[1];
  const float* Wq    = (const float*)d_in[2];
  const float* Wk    = (const float*)d_in[3];
  const float* Wv    = (const float*)d_in[4];
  float*       out   = (float*)d_out;

  char* w = (char*)d_ws;
  auto take = [&](size_t bytes) -> char* {
    char* p = w;
    w += (bytes + 255) & ~(size_t)255;
    return p;
  };
  __bf16* Xq  = (__bf16*)take((size_t)MPROJ * HIDDEN * 2);
  __bf16* Xk  = (__bf16*)take((size_t)MPROJ * HIDDEN * 2);
  __bf16* Wqb = (__bf16*)take((size_t)HIDDEN * HIDDEN * 2);
  __bf16* Wkb = (__bf16*)take((size_t)HIDDEN * HIDDEN * 2);
  __bf16* Wvb = (__bf16*)take((size_t)HIDDEN * HIDDEN * 2);
  __bf16* Qb  = (__bf16*)take((size_t)MPROJ * HIDDEN * 2);
  __bf16* Kb  = (__bf16*)take((size_t)MPROJ * HIDDEN * 2);
  __bf16* Vt  = (__bf16*)take((size_t)MPROJ * HIDDEN * 2);
  float*  Sc  = (float*)take((size_t)SEQ * SEQ * 4);
  __bf16* Pb  = (__bf16*)take((size_t)SEQ * SEQ * 2);

  // f32 -> bf16 conversions (once)
  {
    const int nX = MPROJ * HIDDEN / 8;    // 1,048,576
    const int nW = HIDDEN * HIDDEN / 8;   // 131,072
    convert_kernel<<<nX / 256, 256, 0, stream>>>(query, Xq, nX);
    convert_kernel<<<nX / 256, 256, 0, stream>>>(key, Xk, nX);
    convert_kernel<<<nW / 256, 256, 0, stream>>>(Wq, Wqb, nW);
    convert_kernel<<<nW / 256, 256, 0, stream>>>(Wk, Wkb, nW);
    convert_kernel<<<nW / 256, 256, 0, stream>>>(Wv, Wvb, nW);
  }

  // projections: M=8192, N=1024, K=1024 -> (8192/128)*(1024/256) = 256 blocks
  gemm_nt_kernel<0><<<256, 256, 0, stream>>>(Xq, Wqb, Qb, HIDDEN, 4, HIDDEN);
  gemm_nt_kernel<0><<<256, 256, 0, stream>>>(Xk, Wkb, Kb, HIDDEN, 4, HIDDEN);
  gemm_nt_kernel<1><<<256, 256, 0, stream>>>(Xk, Wvb, Vt, HIDDEN, 4, 0);

  for (int b = 0; b < BATCH; ++b) {
    const __bf16* Qbb = Qb + (size_t)b * SEQ * HIDDEN;
    const __bf16* Kbb = Kb + (size_t)b * SEQ * HIDDEN;
    const __bf16* Vtb = Vt + (size_t)b * HIDDEN * SEQ;
    float* outb = out + (size_t)b * SEQ * HIDDEN;

    // scores: M=N=2048, K=1024 -> 16*8 = 128 blocks
    gemm_nt_kernel<2><<<128, 256, 0, stream>>>(Qbb, Kbb, Sc, HIDDEN, 8, SEQ);
    softmax_kernel<<<SEQ, 256, 0, stream>>>(Sc, Pb);
    // out = P @ V: M=2048, N=1024, K=2048 -> 16*4 = 64 blocks
    gemm_nt_kernel<2><<<64, 256, 0, stream>>>(Pb, Vtb, outb, SEQ, 4, HIDDEN);
  }
}
